// MultiHeadAttention_10144712753523
// MI455X (gfx1250) — matrix operand order repaired and checked by reading, not compiled, not run on hardware
//
#include <hip/hip_runtime.h>

// ---------------------------------------------------------------------------
// MHA for MI455X (gfx1250, wave32).  bf16 WMMA (v_wmma_f32_16x16x32_bf16),
// flash-attention streaming softmax (never materializes the 512MB score
// tensor), LDS staging of K/V tiles, all intermediates bf16 in d_ws (32MB,
// fits in the 192MB L2 between kernels).
// ---------------------------------------------------------------------------

typedef __attribute__((ext_vector_type(16))) __bf16 v16bf;
typedef __attribute__((ext_vector_type(8)))  float  v8f;
typedef __bf16 bf16;

#define D_MODEL 1024
#define N_HEAD  16
#define DKH     64
#define BATCH   2
#define SEQ     2048
#define MROWS   (BATCH * SEQ)
#define NEGVAL  (-1.0e9f)

static __device__ __forceinline__ v8f wmma_bf16(v16bf a, v16bf b, v8f c) {
  return __builtin_amdgcn_wmma_f32_16x16x32_bf16(
      /*neg_a=*/false, a, /*neg_b=*/false, b,
      /*c_mod=*/(short)0, c, /*reuse_a=*/false, /*reuse_b=*/false);
}

static __device__ __forceinline__ v8f zero8() {
  v8f z = {0.f, 0.f, 0.f, 0.f, 0.f, 0.f, 0.f, 0.f};
  return z;
}

// ---------------------------------------------------------------------------
// Projection GEMM: Cbf16[M x 1024] = A_f32[M x 1024] @ W_f32[1024 x 1024]
// Block: 256 thr (8 waves), block tile 128(M) x 64(N), wave tile 32 x 32.
// scale folded into the store (used to pre-scale Q by 1/sqrt(d_k)).
// ---------------------------------------------------------------------------
__global__ __launch_bounds__(256) void proj_gemm_kernel(
    const float* __restrict__ A, const float* __restrict__ W,
    bf16* __restrict__ C, float scale) {
  const int lane = threadIdx.x & 31;
  const int wid  = threadIdx.x >> 5;
  const int half = lane >> 4;
  const int ln   = lane & 15;
  const int row0 = blockIdx.x * 128 + (wid >> 1) * 32;
  const int col0 = blockIdx.y * 64  + (wid & 1) * 32;

  v8f acc00 = zero8(), acc01 = zero8(), acc10 = zero8(), acc11 = zero8();

  for (int k0 = 0; k0 < D_MODEL; k0 += 32) {
    v16bf a0, a1, b0, b1;
    // A fragments: 16x32 bf16, lane holds row, K = e + 16*half
    {
      const float* p0 = A + (size_t)(row0 + ln) * D_MODEL + k0 + 16 * half;
      const float* p1 = A + (size_t)(row0 + 16 + ln) * D_MODEL + k0 + 16 * half;
#pragma unroll
      for (int e = 0; e < 8; ++e) {
        a0[e]     = (bf16)p0[e];
        a0[8 + e] = (bf16)p0[8 + e];
        a1[e]     = (bf16)p1[e];
        a1[8 + e] = (bf16)p1[8 + e];
      }
    }
    // B fragments: 32x16 bf16, lane holds column col0+ln, K = e + 16*half
    {
      const float* wp = W + (size_t)(k0 + 16 * half) * D_MODEL + col0 + ln;
#pragma unroll
      for (int e = 0; e < 16; ++e) {
        b0[e] = (bf16)wp[(size_t)e * D_MODEL];
        b1[e] = (bf16)wp[(size_t)e * D_MODEL + 16];
      }
    }
    acc00 = wmma_bf16(a0, b0, acc00);
    acc01 = wmma_bf16(a0, b1, acc01);
    acc10 = wmma_bf16(a1, b0, acc10);
    acc11 = wmma_bf16(a1, b1, acc11);
  }

#pragma unroll
  for (int r = 0; r < 8; ++r) {
    const int rr = row0 + r + 8 * half;
    C[(size_t)rr * D_MODEL + col0 + ln]            = (bf16)(acc00[r] * scale);
    C[(size_t)rr * D_MODEL + col0 + 16 + ln]       = (bf16)(acc01[r] * scale);
    C[(size_t)(rr + 16) * D_MODEL + col0 + ln]     = (bf16)(acc10[r] * scale);
    C[(size_t)(rr + 16) * D_MODEL + col0 + 16 + ln] = (bf16)(acc11[r] * scale);
  }
}

// ---------------------------------------------------------------------------
// Output GEMM: Out_f32[M x 1024] = A_bf16[M x 1024] @ W_f32[1024 x 1024] + b
// ---------------------------------------------------------------------------
__global__ __launch_bounds__(256) void out_gemm_kernel(
    const bf16* __restrict__ A, const float* __restrict__ W,
    const float* __restrict__ bias, float* __restrict__ C) {
  const int lane = threadIdx.x & 31;
  const int wid  = threadIdx.x >> 5;
  const int half = lane >> 4;
  const int ln   = lane & 15;
  const int row0 = blockIdx.x * 128 + (wid >> 1) * 32;
  const int col0 = blockIdx.y * 64  + (wid & 1) * 32;

  v8f acc00 = zero8(), acc01 = zero8(), acc10 = zero8(), acc11 = zero8();

  for (int k0 = 0; k0 < D_MODEL; k0 += 32) {
    v16bf a0, a1, b0, b1;
    {
      const bf16* p0 = A + (size_t)(row0 + ln) * D_MODEL + k0 + 16 * half;
      const bf16* p1 = A + (size_t)(row0 + 16 + ln) * D_MODEL + k0 + 16 * half;
#pragma unroll
      for (int e = 0; e < 8; ++e) {
        a0[e]     = p0[e];
        a0[8 + e] = p0[8 + e];
        a1[e]     = p1[e];
        a1[8 + e] = p1[8 + e];
      }
    }
    {
      const float* wp = W + (size_t)(k0 + 16 * half) * D_MODEL + col0 + ln;
#pragma unroll
      for (int e = 0; e < 16; ++e) {
        b0[e] = (bf16)wp[(size_t)e * D_MODEL];
        b1[e] = (bf16)wp[(size_t)e * D_MODEL + 16];
      }
    }
    acc00 = wmma_bf16(a0, b0, acc00);
    acc01 = wmma_bf16(a0, b1, acc01);
    acc10 = wmma_bf16(a1, b0, acc10);
    acc11 = wmma_bf16(a1, b1, acc11);
  }

  const float bia0 = bias[col0 + ln];
  const float bia1 = bias[col0 + 16 + ln];
#pragma unroll
  for (int r = 0; r < 8; ++r) {
    const int rr = row0 + r + 8 * half;
    C[(size_t)rr * D_MODEL + col0 + ln]             = acc00[r] + bia0;
    C[(size_t)rr * D_MODEL + col0 + 16 + ln]        = acc01[r] + bia1;
    C[(size_t)(rr + 16) * D_MODEL + col0 + ln]      = acc10[r] + bia0;
    C[(size_t)(rr + 16) * D_MODEL + col0 + 16 + ln] = acc11[r] + bia1;
  }
}

// ---------------------------------------------------------------------------
// Flash attention. One wave = 16 query rows of one (b,h); 8 waves per block
// share (b,h) so K/V 32x64 tiles are staged cooperatively in LDS.
// Q is pre-scaled by 1/sqrt(d_k).  Online softmax with per-row running
// (max, sum); row statistics live per-vgpr-slot, reduced across the 16-lane
// half that owns each row (C layout: row = vgpr + 8*half, col = lane&15).
// ---------------------------------------------------------------------------
__global__ __launch_bounds__(256) void attn_kernel(
    const bf16* __restrict__ Q, const bf16* __restrict__ K,
    const bf16* __restrict__ V, const int* __restrict__ mask,
    bf16* __restrict__ O) {
  __shared__ __align__(16) bf16 kt_s[32 * 64];
  __shared__ __align__(16) bf16 vt_s[32 * 64];
  __shared__ float pb_s[8][16 * 32];

  const int lane = threadIdx.x & 31;
  const int wid  = threadIdx.x >> 5;
  const int half = lane >> 4;
  const int ln   = lane & 15;

  const int bh   = blockIdx.x >> 4;  // 0..31  (b*16 + h)
  const int qblk = blockIdx.x & 15;  // 0..15
  const int b    = bh >> 4;
  const int h    = bh & 15;
  const int q0   = (qblk * 8 + wid) * 16;  // query tile start in [0, SEQ)
  const size_t rowbase = (size_t)b * SEQ;
  const int h64 = h * DKH;

  // Q fragments (16 rows x 64 dims -> two 16x32 A fragments), loaded once.
  v16bf aq0, aq1;
  {
    const bf16* p = Q + (rowbase + q0 + ln) * D_MODEL + h64 + 16 * half;
#pragma unroll
    for (int e = 0; e < 8; ++e) {
      aq0[e]     = p[e];        // d = 16*half + e
      aq0[8 + e] = p[8 + e];    // d = 8 + 16*half + e
      aq1[e]     = p[32 + e];   // d = 32 + 16*half + e
      aq1[8 + e] = p[40 + e];   // d = 40 + 16*half + e
    }
  }

  float mcur[8], lsum[8], alpha[8];
#pragma unroll
  for (int r = 0; r < 8; ++r) { mcur[r] = -1.0e30f; lsum[r] = 0.0f; }
  v8f oacc[4] = {zero8(), zero8(), zero8(), zero8()};

  const int tcr = threadIdx.x >> 3;        // tile row   0..31
  const int tcc = (threadIdx.x & 7) * 8;   // tile col chunk (8 bf16 = 16B)

  for (int kb = 0; kb < SEQ; kb += 32) {
    // Cooperative K/V tile loads: 256 threads x one b128 each per tile.
    *(uint4*)&kt_s[tcr * 64 + tcc] =
        *(const uint4*)&K[(rowbase + kb + tcr) * D_MODEL + h64 + tcc];
    *(uint4*)&vt_s[tcr * 64 + tcc] =
        *(const uint4*)&V[(rowbase + kb + tcr) * D_MODEL + h64 + tcc];
    __syncthreads();

    // Scores: S0 = Q(16x64) . K[kb..kb+15]^T, S1 = Q . K[kb+16..kb+31]^T
    v8f s0 = zero8(), s1 = zero8();
    {
      v16bf bk;
      const bf16* pk0 = &kt_s[(0 + ln) * 64 + 16 * half];   // keys kb..kb+15
      const bf16* pk1 = &kt_s[(16 + ln) * 64 + 16 * half];  // keys kb+16..+31
#pragma unroll
      for (int e = 0; e < 16; ++e) bk[e] = pk0[e];          // d = 0..31
      s0 = wmma_bf16(aq0, bk, s0);
#pragma unroll
      for (int e = 0; e < 16; ++e) bk[e] = pk0[32 + e];     // d = 32..63
      s0 = wmma_bf16(aq1, bk, s0);
#pragma unroll
      for (int e = 0; e < 16; ++e) bk[e] = pk1[e];
      s1 = wmma_bf16(aq0, bk, s1);
#pragma unroll
      for (int e = 0; e < 16; ++e) bk[e] = pk1[32 + e];
      s1 = wmma_bf16(aq1, bk, s1);
    }

    // Mask + online softmax update (per-row stats in vgpr slot r).
    const int* mp = mask + (size_t)(q0 + 8 * half) * SEQ + kb + ln;
#pragma unroll
    for (int r = 0; r < 8; ++r) {
      const int m0 = mp[(size_t)r * SEQ];
      const int m1 = mp[(size_t)r * SEQ + 16];
      float v0 = m0 ? s0[r] : NEGVAL;
      float v1 = m1 ? s1[r] : NEGVAL;
      float mx = fmaxf(v0, v1);
      mx = fmaxf(mx, __shfl_xor(mx, 1, 32));
      mx = fmaxf(mx, __shfl_xor(mx, 2, 32));
      mx = fmaxf(mx, __shfl_xor(mx, 4, 32));
      mx = fmaxf(mx, __shfl_xor(mx, 8, 32));
      const float mnew = fmaxf(mcur[r], mx);
      const float al = __expf(mcur[r] - mnew);
      const float p0 = __expf(v0 - mnew);
      const float p1 = __expf(v1 - mnew);
      float ps = p0 + p1;
      ps += __shfl_xor(ps, 1, 32);
      ps += __shfl_xor(ps, 2, 32);
      ps += __shfl_xor(ps, 4, 32);
      ps += __shfl_xor(ps, 8, 32);
      lsum[r] = lsum[r] * al + ps;
      mcur[r] = mnew;
      alpha[r] = al;
      s0[r] = p0;
      s1[r] = p1;
    }
#pragma unroll
    for (int r = 0; r < 8; ++r) {
      oacc[0][r] *= alpha[r];
      oacc[1][r] *= alpha[r];
      oacc[2][r] *= alpha[r];
      oacc[3][r] *= alpha[r];
    }

    // P (C layout) -> per-wave LDS -> A fragment (layout transpose).
    float* pb = pb_s[wid];
#pragma unroll
    for (int r = 0; r < 8; ++r) {
      pb[(r + 8 * half) * 32 + ln]      = s0[r];
      pb[(r + 8 * half) * 32 + 16 + ln] = s1[r];
    }
    __asm__ volatile("" ::: "memory");  // keep ds_store before ds_load; HW is
                                        // in-order per wave for LDS ops.
    v16bf pa;
    {
      const float* pp = pb + ln * 32 + 16 * half;
#pragma unroll
      for (int e = 0; e < 8; ++e) {
        pa[e]     = (bf16)pp[e];
        pa[8 + e] = (bf16)pp[8 + e];
      }
    }

    // O += P(16x32) . V(32x64)  -> 4 WMMAs across the 4 d-column tiles.
    {
      v16bf bv;
#pragma unroll
      for (int j = 0; j < 4; ++j) {
#pragma unroll
        for (int e = 0; e < 16; ++e)
          bv[e] = vt_s[(e + 16 * half) * 64 + j * 16 + ln];
        oacc[j] = wmma_bf16(pa, bv, oacc[j]);
      }
    }
    __syncthreads();  // tiles consumed; safe to overwrite next iteration
  }

  // Normalize and store concat output (bf16).
#pragma unroll
  for (int r = 0; r < 8; ++r) {
    const float inv = 1.0f / lsum[r];
    const size_t orow = (rowbase + q0 + r + 8 * half) * D_MODEL + h64;
    O[orow + ln]      = (bf16)(oacc[0][r] * inv);
    O[orow + 16 + ln] = (bf16)(oacc[1][r] * inv);
    O[orow + 32 + ln] = (bf16)(oacc[2][r] * inv);
    O[orow + 48 + ln] = (bf16)(oacc[3][r] * inv);
  }
}

// ---------------------------------------------------------------------------
extern "C" void kernel_launch(void* const* d_in, const int* in_sizes, int n_in,
                              void* d_out, int out_size, void* d_ws,
                              size_t ws_size, hipStream_t stream) {
  (void)in_sizes; (void)n_in; (void)out_size; (void)ws_size;
  const float* xq = (const float*)d_in[0];
  const float* xk = (const float*)d_in[1];
  const float* xv = (const float*)d_in[2];
  const int*   mk = (const int*)d_in[3];
  const float* wq = (const float*)d_in[4];
  const float* wk = (const float*)d_in[5];
  const float* wv = (const float*)d_in[6];
  const float* wo = (const float*)d_in[7];
  const float* bo = (const float*)d_in[8];

  // Workspace: Q, K, V, attn-out  (each 4M bf16 = 8MB; 32MB total).
  bf16* Qb = (bf16*)d_ws;
  bf16* Kb = Qb + (size_t)MROWS * D_MODEL;
  bf16* Vb = Kb + (size_t)MROWS * D_MODEL;
  bf16* Ob = Vb + (size_t)MROWS * D_MODEL;

  dim3 blk(256, 1, 1);
  dim3 gp(MROWS / 128, D_MODEL / 64, 1);  // 32 x 16 blocks

  // scores get /sqrt(64): fold 0.125 into Q.
  proj_gemm_kernel<<<gp, blk, 0, stream>>>(xq, wq, Qb, 0.125f);
  proj_gemm_kernel<<<gp, blk, 0, stream>>>(xk, wk, Kb, 1.0f);
  proj_gemm_kernel<<<gp, blk, 0, stream>>>(xv, wv, Vb, 1.0f);

  const int nblk = BATCH * N_HEAD * (SEQ / 16 / 8);  // 512
  attn_kernel<<<dim3(nblk, 1, 1), blk, 0, stream>>>(Qb, Kb, Vb, mk, Ob);

  out_gemm_kernel<<<gp, blk, 0, stream>>>(Ob, wo, bo, (float*)d_out);
}
